// MultiFieldEncoder_3633542333053
// MI455X (gfx1250) — matrix operand order repaired and checked by reading, not compiled, not run on hardware
//
#include <hip/hip_runtime.h>

typedef __attribute__((ext_vector_type(16))) _Float16 v16h;
typedef __attribute__((ext_vector_type(8)))  float    v8f;
typedef __attribute__((ext_vector_type(4)))  float    v4f;

// Problem constants (match reference)
constexpr int Bc  = 2;
constexpr int Nn  = 100000;
constexpr int NEe = 800000;
constexpr int Fd  = 2;
constexpr int Dd  = 128;
constexpr int TDd = 11;
constexpr int CDd = 32;
constexpr int SDd = 10;

__device__ __forceinline__ float silu_f(float x) {
    // fast silu: one v_exp_f32 + one v_rcp_f32 (TRANS ops, co-issue with VALU)
    return x * __builtin_amdgcn_rcpf(1.0f + __expf(-x));
}

// ---------------------------------------------------------------------------
// Kernel 1: FiLM parameter MLPs (tiny). One block, 256 threads: b = t>>7,
// d = t&127. Produces gamma[2][128] then beta[2][128] into ws.
// ---------------------------------------------------------------------------
__global__ __launch_bounds__(256) void film_kernel(
    const float* __restrict__ time_i, const float* __restrict__ cond_i,
    const float* __restrict__ spat_i,
    const float* __restrict__ tw1, const float* __restrict__ tb1,
    const float* __restrict__ tw2, const float* __restrict__ tb2,
    const float* __restrict__ cw1, const float* __restrict__ cb1,
    const float* __restrict__ cw2, const float* __restrict__ cb2,
    const float* __restrict__ sw1, const float* __restrict__ sb1,
    const float* __restrict__ sw2, const float* __restrict__ sb2,
    const float* __restrict__ fw1, const float* __restrict__ fb1,
    const float* __restrict__ fw2, const float* __restrict__ fb2,
    float* __restrict__ gb)
{
    __shared__ float sh_hid[2][128];
    __shared__ float sh_enc[2][384];  // [cond | time | spat]

    const int t = threadIdx.x;
    const int b = t >> 7;
    const int d = t & 127;

    auto stage = [&](const float* __restrict__ x, int in,
                     const float* __restrict__ w1_, const float* __restrict__ b1_,
                     const float* __restrict__ w2_, const float* __restrict__ b2_,
                     int off) {
        __syncthreads();
        float acc = b1_[d];
        for (int k = 0; k < in; ++k)
            acc += x[b * in + k] * w1_[k * 128 + d];
        sh_hid[b][d] = silu_f(acc);
        __syncthreads();
        float acc2 = b2_[d];
        for (int k = 0; k < 128; ++k)
            acc2 += sh_hid[b][k] * w2_[k * 128 + d];
        sh_enc[b][off + d] = acc2;
    };

    stage(cond_i, CDd, cw1, cb1, cw2, cb2, 0);
    stage(time_i, TDd, tw1, tb1, tw2, tb2, 128);
    stage(spat_i, SDd, sw1, sb1, sw2, sb2, 256);
    __syncthreads();

    // fuse: 384 -> 128 (SiLU) -> 256
    float acc = fb1[d];
    for (int k = 0; k < 384; ++k)
        acc += sh_enc[b][k] * fw1[k * 128 + d];
    sh_hid[b][d] = silu_f(acc);
    __syncthreads();

    float accg = fb2[d];
    float accb = fb2[128 + d];
    for (int k = 0; k < 128; ++k) {
        float hv = sh_hid[b][k];
        accg += hv * fw2[k * 256 + d];
        accb += hv * fw2[k * 256 + 128 + d];
    }
    gb[b * 128 + d]       = accg;   // gamma
    gb[256 + b * 128 + d] = accb;   // beta
}

// ---------------------------------------------------------------------------
// Kernel 2: per-field node MLP + FiLM.  block = 256 thr = 8 waves, each wave
// owns 16-node tiles (grid-stride).  fields_w2[f] converted to f16 in LDS
// (pre-swizzled into the wave32 WMMA B-fragment layout); output tile staged
// through LDS so global stores are 512B coalesced non-temporal b128 bursts.
// ---------------------------------------------------------------------------
__global__ __launch_bounds__(256) void node_film_kernel(
    const float* __restrict__ node_pos, const float* __restrict__ state_in,
    const float* __restrict__ w1, const float* __restrict__ b1,
    const float* __restrict__ w2, const float* __restrict__ b2,
    const float* __restrict__ gb, float* __restrict__ V_out)
{
    __shared__ __align__(32) _Float16 shB[4][8][32][16]; // [kchunk][ntile][lane][elem]
    __shared__ float shW1[4 * 128];
    __shared__ float shB1[128];
    __shared__ float shGam[128];   // gamma
    __shared__ float shBet[128];   // b2*gamma + beta (folded epilogue)
    __shared__ float shOut[8][16][132]; // per-wave 16x128 tile, padded stride

    const int f = blockIdx.y;
    const int b = blockIdx.z;
    const int t = threadIdx.x;

    // ---- stage weights into LDS (B-fragment swizzle: K = kc*32 + 16*hs + i,
    //      col = nt*16 + (lane&15)) ----
    const float* w2f = w2 + f * Dd * Dd;
#pragma unroll 8
    for (int idx = t; idx < 4 * 8 * 32 * 16; idx += 256) {
        int i    = idx & 15;
        int lane = (idx >> 4) & 31;
        int nt   = (idx >> 9) & 7;
        int kc   = idx >> 12;
        int K    = kc * 32 + ((lane >> 4) * 16) + i;
        int col  = nt * 16 + (lane & 15);
        shB[kc][nt][lane][i] = (_Float16)w2f[K * Dd + col];
    }
    for (int idx = t; idx < 512; idx += 256) shW1[idx] = w1[f * 4 * Dd + idx];
    if (t < 128) {
        shB1[t] = b1[f * Dd + t];
        float g = gb[b * Dd + t];
        shGam[t] = g;
        shBet[t] = b2[f * Dd + t] * g + gb[2 * Dd + b * Dd + t];
    }
    __syncthreads();

    const int wave = t >> 5;
    const int lane = t & 31;
    const int hs   = lane >> 4;   // half-select
    const int m    = lane & 15;   // row / col within tile

    const int ntiles = (Nn + 15) / 16;
    for (int tile = blockIdx.x * 8 + wave; tile < ntiles; tile += gridDim.x * 8) {
        const int tile0 = tile * 16;
        const int node  = tile0 + m;
        const int nc    = node < Nn ? node : Nn - 1;
        const size_t pb = (size_t)b * Nn + nc;
        const float x0 = node_pos[pb * 3 + 0];
        const float x1 = node_pos[pb * 3 + 1];
        const float x2 = node_pos[pb * 3 + 2];
        const float x3 = state_in[pb * Fd + f];

        // Hidden layer: each lane computes exactly the 64 K-values its
        // A-fragment needs (contiguous k = 16*hs + i per chunk, same order as shB).
        v16h a[4];
#pragma unroll
        for (int kc = 0; kc < 4; ++kc) {
#pragma unroll
            for (int i = 0; i < 16; ++i) {
                int Kl = hs * 16 + i;
                int k  = kc * 32 + Kl;
                float acc = shB1[k] + x0 * shW1[0 * 128 + k] + x1 * shW1[1 * 128 + k]
                                    + x2 * shW1[2 * 128 + k] + x3 * shW1[3 * 128 + k];
                a[kc][i] = (_Float16)silu_f(acc);
            }
        }

        // 8 N-tiles in two (rolled!) halves of 4: bounds live accumulators to
        // 4 chains; per K-chunk all 8 ds_load_b128 issue before the 4 WMMAs.
#pragma unroll 1
        for (int half = 0; half < 2; ++half) {
            v8f c[4] = {};
#pragma unroll
            for (int kc = 0; kc < 4; ++kc) {
                v16h bf[4];
#pragma unroll
                for (int q = 0; q < 4; ++q)
                    bf[q] = *(const v16h*)(&shB[kc][half * 4 + q][lane][0]);
#pragma unroll
                for (int q = 0; q < 4; ++q)
                    c[q] = __builtin_amdgcn_wmma_f32_16x16x32_f16(
                               false, a[kc], false, bf[q], (short)0, c[q], false, false);
            }
#pragma unroll
            for (int q = 0; q < 4; ++q) {
                const int col = (half * 4 + q) * 16 + m;
                const float gam = shGam[col];
                const float bet = shBet[col];
#pragma unroll
                for (int r = 0; r < 8; ++r) {
                    shOut[wave][r + 8 * hs][col] = c[q][r] * gam + bet;
                }
            }
        }

        // Coalesced writeback: one 512B non-temporal burst per node row.
#pragma unroll 4
        for (int rr = 0; rr < 16; ++rr) {
            const int nd = tile0 + rr;
            if (nd < Nn) {
                v4f v = *(const v4f*)(&shOut[wave][rr][lane * 4]);
                __builtin_nontemporal_store(
                    v, (v4f*)(&V_out[(((size_t)b * Nn + nd) * Fd + f) * Dd + lane * 4]));
            }
        }
    }
}

// ---------------------------------------------------------------------------
// Kernel 3: edge MLP.  Same WMMA structure; features gathered via edges.
// ---------------------------------------------------------------------------
__global__ __launch_bounds__(256) void edge_kernel(
    const float* __restrict__ node_pos, const int* __restrict__ edges,
    const float* __restrict__ fw1, const float* __restrict__ fb1,
    const float* __restrict__ fw2, const float* __restrict__ fb2,
    float* __restrict__ E_out)
{
    __shared__ __align__(32) _Float16 shB[4][8][32][16];
    __shared__ float shW1[7 * 128];
    __shared__ float shB1[128];
    __shared__ float shB2[128];
    __shared__ float shOut[8][16][132];

    const int b = blockIdx.y;
    const int t = threadIdx.x;

#pragma unroll 8
    for (int idx = t; idx < 4 * 8 * 32 * 16; idx += 256) {
        int i    = idx & 15;
        int lane = (idx >> 4) & 31;
        int nt   = (idx >> 9) & 7;
        int kc   = idx >> 12;
        int K    = kc * 32 + ((lane >> 4) * 16) + i;
        int col  = nt * 16 + (lane & 15);
        shB[kc][nt][lane][i] = (_Float16)fw2[K * Dd + col];
    }
    for (int idx = t; idx < 7 * 128; idx += 256) shW1[idx] = fw1[idx];
    if (t < 128) { shB1[t] = fb1[t]; shB2[t] = fb2[t]; }
    __syncthreads();

    const int wave = t >> 5;
    const int lane = t & 31;
    const int hs   = lane >> 4;
    const int m    = lane & 15;

    const int ntiles = (NEe + 15) / 16;
    for (int tile = blockIdx.x * 8 + wave; tile < ntiles; tile += gridDim.x * 8) {
        const int tile0 = tile * 16;
        const int e     = tile0 + m;
        const int ec    = e < NEe ? e : NEe - 1;
        const size_t eb = ((size_t)b * NEe + ec) * 2;
        const int si = edges[eb + 0];
        const int ri = edges[eb + 1];
        const size_t sp = ((size_t)b * Nn + si) * 3;
        const size_t rp = ((size_t)b * Nn + ri) * 3;
        const float d0 = node_pos[rp + 0] - node_pos[sp + 0];
        const float d1 = node_pos[rp + 1] - node_pos[sp + 1];
        const float d2 = node_pos[rp + 2] - node_pos[sp + 2];
        const float nrm = sqrtf(d0 * d0 + d1 * d1 + d2 * d2 + 1e-8f);
        float feat[7] = {d0, d1, d2, -d0, -d1, -d2, nrm};

        v16h a[4];
#pragma unroll
        for (int kc = 0; kc < 4; ++kc) {
#pragma unroll
            for (int i = 0; i < 16; ++i) {
                int Kl = hs * 16 + i;
                int k  = kc * 32 + Kl;
                float acc = shB1[k];
#pragma unroll
                for (int j = 0; j < 7; ++j) acc += feat[j] * shW1[j * 128 + k];
                a[kc][i] = (_Float16)silu_f(acc);
            }
        }

#pragma unroll 1
        for (int half = 0; half < 2; ++half) {
            v8f c[4] = {};
#pragma unroll
            for (int kc = 0; kc < 4; ++kc) {
                v16h bf[4];
#pragma unroll
                for (int q = 0; q < 4; ++q)
                    bf[q] = *(const v16h*)(&shB[kc][half * 4 + q][lane][0]);
#pragma unroll
                for (int q = 0; q < 4; ++q)
                    c[q] = __builtin_amdgcn_wmma_f32_16x16x32_f16(
                               false, a[kc], false, bf[q], (short)0, c[q], false, false);
            }
#pragma unroll
            for (int q = 0; q < 4; ++q) {
                const int col = (half * 4 + q) * 16 + m;
                const float add = shB2[col];
#pragma unroll
                for (int r = 0; r < 8; ++r) {
                    shOut[wave][r + 8 * hs][col] = c[q][r] + add;
                }
            }
        }

#pragma unroll 4
        for (int rr = 0; rr < 16; ++rr) {
            const int ed = tile0 + rr;
            if (ed < NEe) {
                v4f v = *(const v4f*)(&shOut[wave][rr][lane * 4]);
                __builtin_nontemporal_store(
                    v, (v4f*)(&E_out[((size_t)b * NEe + ed) * Dd + lane * 4]));
            }
        }
    }
}

// ---------------------------------------------------------------------------
extern "C" void kernel_launch(void* const* d_in, const int* in_sizes, int n_in,
                              void* d_out, int out_size, void* d_ws, size_t ws_size,
                              hipStream_t stream) {
    (void)in_sizes; (void)n_in; (void)out_size; (void)ws_size;

    const float* node_pos  = (const float*)d_in[0];
    const float* state_in  = (const float*)d_in[1];
    const float* time_i    = (const float*)d_in[2];
    const float* cond_i    = (const float*)d_in[3];
    const float* spat_i    = (const float*)d_in[4];
    const int*   edges     = (const int*)  d_in[5];
    const float* fields_w1 = (const float*)d_in[6];
    const float* fields_b1 = (const float*)d_in[7];
    const float* fields_w2 = (const float*)d_in[8];
    const float* fields_b2 = (const float*)d_in[9];
    const float* tw1 = (const float*)d_in[10]; const float* tb1 = (const float*)d_in[11];
    const float* tw2 = (const float*)d_in[12]; const float* tb2 = (const float*)d_in[13];
    const float* cw1 = (const float*)d_in[14]; const float* cb1 = (const float*)d_in[15];
    const float* cw2 = (const float*)d_in[16]; const float* cb2 = (const float*)d_in[17];
    const float* sw1 = (const float*)d_in[18]; const float* sb1 = (const float*)d_in[19];
    const float* sw2 = (const float*)d_in[20]; const float* sb2 = (const float*)d_in[21];
    const float* fuw1 = (const float*)d_in[22]; const float* fub1 = (const float*)d_in[23];
    const float* fuw2 = (const float*)d_in[24]; const float* fub2 = (const float*)d_in[25];
    const float* few1 = (const float*)d_in[26]; const float* feb1 = (const float*)d_in[27];
    const float* few2 = (const float*)d_in[28]; const float* feb2 = (const float*)d_in[29];

    float* out   = (float*)d_out;
    float* gb    = (float*)d_ws;                       // gamma[2][128], beta[2][128]
    float* V_out = out;                                // [B,N,F,D]
    float* E_out = out + (size_t)Bc * Nn * Fd * Dd;    // [B,NE,D]

    film_kernel<<<1, 256, 0, stream>>>(
        time_i, cond_i, spat_i,
        tw1, tb1, tw2, tb2,
        cw1, cb1, cw2, cb2,
        sw1, sb1, sw2, sb2,
        fuw1, fub1, fuw2, fub2, gb);

    node_film_kernel<<<dim3(160, Fd, Bc), 256, 0, stream>>>(
        node_pos, state_in, fields_w1, fields_b1, fields_w2, fields_b2, gb, V_out);

    edge_kernel<<<dim3(512, Bc), 256, 0, stream>>>(
        node_pos, edges, few1, feb1, few2, feb2, E_out);
}
